// NLinear_5669356831822
// MI455X (gfx1250) — matrix operand order repaired and checked by reading, not compiled, not run on hardware
//
#include <hip/hip_runtime.h>
#include <hip/hip_bf16.h>

#define N_MAT 100
#define D_IN  384
#define D_OUT 384
#define BATCH 1024

#define BM 128
#define BN 128
#define BK 32
#define KT (D_IN / BK)   // 12 k-iterations
#define LDT 40   // padded LDS row stride in bf16 elems (80B = 20 words: 16B aligned, conflict-free for 16 rows)

typedef __attribute__((ext_vector_type(16))) __bf16 v16bf;
typedef __attribute__((ext_vector_type(8)))  __bf16 v8bf;
typedef __attribute__((ext_vector_type(8)))  float  v8f;
typedef __attribute__((ext_vector_type(4)))  float  v4f;

__device__ __forceinline__ v16bf cat8(v8bf a, v8bf b) {
  return __builtin_shufflevector(a, b, 0,1,2,3,4,5,6,7,8,9,10,11,12,13,14,15);
}

__global__ __launch_bounds__(256)
void nlinear_bf16x3_kernel(const float* __restrict__ x,
                           const float* __restrict__ w,
                           const float* __restrict__ bias,
                           float* __restrict__ out) {
  // [buf][0]=A_hi [buf][1]=A_lo [buf][2]=Bt_hi [buf][3]=Bt_lo  -> 80 KB total
  __shared__ __align__(16) __bf16 lds[2][4][BM * LDT];

  const int tid   = threadIdx.x;
  const int nIdx  = blockIdx.z;        // which of the 100 matrices
  const int mbase = blockIdx.y * BM;   // batch-row tile base
  const int nbase = blockIdx.x * BN;   // d_out tile base

  const int lane    = tid & 31;
  const int wave    = tid >> 5;
  const int halfSel = lane >> 4;       // 0: lanes 0-15, 1: lanes 16-31
  const int ln      = lane & 15;
  const int wm      = wave & 1;        // 2 wave-rows of 64
  const int wn      = wave >> 1;       // 4 wave-cols of 32

  const size_t xRowStride = (size_t)N_MAT * D_IN;   // 38400
  const size_t oRowStride = (size_t)N_MAT * D_OUT;  // 38400

  // ---- loop-invariant per-thread staging coordinates & base pointers
  // A tile: 128 rows x 8 float4 along k ; thread handles rows tid>>3 (+32/it), quad tid&7
  const int aRow = tid >> 3;           // 0..31, +32 per it
  const int aQ   = tid & 7;            // float4 index along k
  // B tile: 32 k-rows x 32 float4 along d_out ; thread handles k-row tid>>5 (+8/it), quad tid&31
  const int bRow = tid >> 5;           // 0..7, +8 per it
  const int bQ   = tid & 31;           // float4 index along d_out

  const float* aPtr[4];
  const float* bPtr[4];
#pragma unroll
  for (int it = 0; it < 4; ++it) {
    aPtr[it] = x + (size_t)nIdx * D_IN
                 + (size_t)(mbase + aRow + it * 32) * xRowStride + aQ * 4;
    bPtr[it] = w + (size_t)nIdx * D_IN * D_OUT
                 + (size_t)(bRow + it * 8) * D_OUT + nbase + bQ * 4;
  }

  v8f acc[4][2];
#pragma unroll
  for (int mi = 0; mi < 4; ++mi)
#pragma unroll
    for (int ni = 0; ni < 2; ++ni)
      acc[mi][ni] = (v8f){0.f,0.f,0.f,0.f,0.f,0.f,0.f,0.f};

  v4f aReg[4], bReg[4];

  // ---- global-load: fetch A (128x32) and W (32x128) f32 tiles into regs (kbase folds to immediates)
  auto loadTiles = [&](int kbase) {
#pragma unroll
    for (int it = 0; it < 4; ++it)
      aReg[it] = *(const v4f*)(aPtr[it] + kbase);                      // contiguous along k
#pragma unroll
    for (int it = 0; it < 4; ++it)
      bReg[it] = *(const v4f*)(bPtr[it] + (size_t)kbase * D_OUT);      // k rows stride D_OUT
  };

  // ---- convert (bf16 hi/lo split, native RNE casts) and store tiles to LDS[buf]
  auto storeTiles = [&](int buf) {
#pragma unroll
    for (int it = 0; it < 4; ++it) {
      const int row = aRow + it * 32;
#pragma unroll
      for (int j = 0; j < 4; ++j) {
        float f  = aReg[it][j];
        __bf16 h = (__bf16)f;
        __bf16 l = (__bf16)(f - (float)h);
        lds[buf][0][row * LDT + aQ * 4 + j] = h;
        lds[buf][1][row * LDT + aQ * 4 + j] = l;
      }
    }
#pragma unroll
    for (int it = 0; it < 4; ++it) {
      const int row = bRow + it * 8;
#pragma unroll
      for (int j = 0; j < 4; ++j) {
        float f  = bReg[it][j];
        __bf16 h = (__bf16)f;
        __bf16 l = (__bf16)(f - (float)h);
        lds[buf][2][(bQ * 4 + j) * LDT + row] = h;   // transposed: Bt[col][k]
        lds[buf][3][(bQ * 4 + j) * LDT + row] = l;
      }
    }
  };

  // prologue: fill buffer 0
  loadTiles(0);
  storeTiles(0);
  __syncthreads();

#pragma unroll
  for (int kt = 0; kt < KT; ++kt) {
    const int buf = kt & 1;
    const bool hasNext = (kt + 1) < KT;

    // issue next tile's global loads early (latency hidden under WMMAs)
    if (hasNext) loadTiles((kt + 1) * BK);

    // ---- fragments from LDS[buf], per documented CDNA5 WMMA VGPR layouts
    // A (16-bit 16x32): lane(r,half): elems 0..7 = K 8h..+7, elems 8..15 = K 16+8h..+7
    v16bf aH[4], aL[4], bH[2], bL[2];
#pragma unroll
    for (int mi = 0; mi < 4; ++mi) {
      int r = (wm * 64 + mi * 16 + ln) * LDT;
      aH[mi] = cat8(*(const v8bf*)&lds[buf][0][r + 8 * halfSel],
                    *(const v8bf*)&lds[buf][0][r + 16 + 8 * halfSel]);
      aL[mi] = cat8(*(const v8bf*)&lds[buf][1][r + 8 * halfSel],
                    *(const v8bf*)&lds[buf][1][r + 16 + 8 * halfSel]);
    }
    // B (16-bit 32x16): lane(half,n): elems 0..7 = K 8h..+7, elems 8..15 = K 16+8h..+7 of column n
#pragma unroll
    for (int ni = 0; ni < 2; ++ni) {
      int r = (wn * 32 + ni * 16 + ln) * LDT + 8 * halfSel;
      bH[ni] = cat8(*(const v8bf*)&lds[buf][2][r],
                    *(const v8bf*)&lds[buf][2][r + 16]);
      bL[ni] = cat8(*(const v8bf*)&lds[buf][3][r],
                    *(const v8bf*)&lds[buf][3][r + 16]);
    }

    // ---- bf16x3 f32 emulation: hi*lo + lo*hi + hi*hi, f32 accumulate
#pragma unroll
    for (int mi = 0; mi < 4; ++mi) {
#pragma unroll
      for (int ni = 0; ni < 2; ++ni) {
        v8f c = acc[mi][ni];
        c = __builtin_amdgcn_wmma_f32_16x16x32_bf16(false, aH[mi], false, bL[ni], (short)0, c, false, false);
        c = __builtin_amdgcn_wmma_f32_16x16x32_bf16(false, aL[mi], false, bH[ni], (short)0, c, false, false);
        c = __builtin_amdgcn_wmma_f32_16x16x32_bf16(false, aH[mi], false, bH[ni], (short)0, c, false, false);
        acc[mi][ni] = c;
      }
    }

    // convert + store next tile into the other buffer (no conflict with reads above)
    if (hasNext) storeTiles(buf ^ 1);

    __syncthreads();   // single barrier per iteration (double-buffered)
  }

  // ---- epilogue: add bias, store f32
  // C/D layout: VGPR v holds row M = v + 8*half, column = lane&15
#pragma unroll
  for (int ni = 0; ni < 2; ++ni) {
    int col = nbase + wn * 32 + ni * 16 + ln;
    float bv = bias[nIdx * D_OUT + col];
#pragma unroll
    for (int mi = 0; mi < 4; ++mi) {
#pragma unroll
      for (int v = 0; v < 8; ++v) {
        int row = mbase + wm * 64 + mi * 16 + halfSel * 8 + v;
        out[(size_t)row * oRowStride + (size_t)nIdx * D_OUT + col] = acc[mi][ni][v] + bv;
      }
    }
  }
}

extern "C" void kernel_launch(void* const* d_in, const int* in_sizes, int n_in,
                              void* d_out, int out_size, void* d_ws, size_t ws_size,
                              hipStream_t stream) {
  const float* x    = (const float*)d_in[0];
  const float* w    = (const float*)d_in[1];
  const float* bias = (const float*)d_in[2];
  float* out        = (float*)d_out;
  (void)in_sizes; (void)n_in; (void)out_size; (void)d_ws; (void)ws_size;

  dim3 grid(D_OUT / BN, BATCH / BM, N_MAT);  // (3, 8, 100)
  dim3 block(256);
  nlinear_bf16x3_kernel<<<grid, block, 0, stream>>>(x, w, bias, out);
}
